// SelfAttention_58171037057536
// MI455X (gfx1250) — matrix operand order repaired and checked by reading, not compiled, not run on hardware
//
#include <hip/hip_runtime.h>
#include <hip/hip_bf16.h>

typedef __attribute__((ext_vector_type(16))) _Float16 v16h;
typedef __attribute__((ext_vector_type(8)))  float    v8f;
typedef unsigned int u32x4 __attribute__((ext_vector_type(4)));
typedef int          i32x4 __attribute__((ext_vector_type(4)));
typedef int          i32x8 __attribute__((ext_vector_type(8)));

#define BATCH 16
#define DMODEL 4096
#define NHEADS 32
#define NKV 8
#define REP 4
#define DH 128
#define SEQ 4096
#define LASTPOS 4095
#define NSPLIT 4
#define SPLIT_LEN 1024
#define QK_SCALE 0.08838834764831845f

// ---- gfx1250 data-movement feature detection -------------------------------
#if defined(__has_builtin)
#  if __has_builtin(__builtin_amdgcn_tensor_load_to_lds)
#    define HAVE_TDM 1
#  endif
#  if __has_builtin(__builtin_amdgcn_global_load_async_to_lds_b128)
#    define HAVE_ASYNC 1
#  endif
#endif
#if defined(__has_include)
#  if __has_include(<hip/amd_detail/amd_gfx1250_TDM.h>)
#    define TDM_6ARG 1
#  endif
#endif

// Tell the compile loop which path was selected (shows up in stderr).
#if HAVE_TDM
#  warning "CDNA5 path: TDM tensor_load_to_lds (TENSORcnt double-buffered)"
#  if TDM_6ARG
#    warning "CDNA5 path: 6-arg TDM builtin (amdgpu-toolchain headers present)"
#  else
#    warning "CDNA5 path: 5-arg TDM builtin (ROCm toolchain)"
#  endif
#elif HAVE_ASYNC
#  warning "CDNA5 path: global_load_async_to_lds_b128 (ASYNCcnt)"
#else
#  warning "CDNA5 path: synchronous VGPR staging fallback"
#endif

__device__ __forceinline__ unsigned lds_offset_of(const void* p) {
  return (unsigned)(unsigned long long)(__attribute__((address_space(3))) const void*)p;
}

#if HAVE_TDM
// Issue one TDM 2D tile load: rows x cols (f32) from global (row stride Nw
// elements) into LDS at lds_byte_off, with optional LDS row padding.
// g1_flags packs data_size/pad config (see CDNA5 ISA 8.4).
__device__ __forceinline__ void tdm_load_2d(const float* gsrc, unsigned lds_byte_off,
                                            int cols, int rows, int stride_elems,
                                            int g1_flags) {
  const unsigned long long ga = (unsigned long long)(const void*)gsrc;
  u32x4 g0;
  g0[0] = 1u;                                  // count=1, user mode, no gather
  g0[1] = lds_byte_off;
  g0[2] = (unsigned)ga;                        // global_addr[31:0]
  g0[3] = (unsigned)(ga >> 32) | 0x80000000u;  // global_addr[56:32] | type=2
  i32x8 g1;
  g1[0] = g1_flags;                 // data_size=4B (+pad config)
  g1[1] = cols << 16;               // tensor_dim0
  g1[2] = rows << 16;               // tensor_dim1
  g1[3] = cols << 16;               // tile_dim0
  g1[4] = rows;                     // tile_dim1 (tile_dim2 = 0)
  g1[5] = stride_elems;             // tensor_dim0_stride[31:0]
  g1[6] = 0;
  g1[7] = 0;
  i32x4 g2 = {0, 0, 0, 0}, g3 = {0, 0, 0, 0};
#if TDM_6ARG
  i32x8 g4 = {0, 0, 0, 0, 0, 0, 0, 0};
  __builtin_amdgcn_tensor_load_to_lds(g0, g1, g2, g3, g4, 0);
#else
  __builtin_amdgcn_tensor_load_to_lds(g0, g1, g2, g3, 0);
#endif
}
#endif

// ---------------------------------------------------------------------------
// Shared 16xN GEMM tile: acc(16x16) += A(16 x DMODEL) * W(DMODEL x Nw)[:,n0:n0+16]
// TDM path: double-buffered 128x16 weight tiles in LDS (row stride 17 floats,
// produced by TDM pad: 1 dword after every 16 dwords -> bank-conflict-free).
// ---------------------------------------------------------------------------
__device__ __forceinline__ v8f tile_gemm16(const float* __restrict__ A,
                                           const float* __restrict__ w,
                                           int Nw, int n0, int lane) {
  const int m  = lane & 15;
  const int hi = lane >> 4;
  v8f acc = {};
#if HAVE_TDM
  __shared__ float wt[2][128 * 17];  // 2 x 8.5KB weight tiles
  const unsigned wt_base = lds_offset_of(&wt[0][0]);
  // data_size=4B | pad_enable | pad_interval=16dw(code 3) | pad_amount=1dw(code 0)
  const int WFLAGS = 0x00020000 | (1 << 20) | (3 << 22);
  tdm_load_2d(w + n0, wt_base, 16, 128, Nw, WFLAGS);
  for (int k0 = 0; k0 < DMODEL; k0 += 128) {
    const int buf = (k0 >> 7) & 1;
    if (k0 + 128 < DMODEL) {
      tdm_load_2d(w + (size_t)(k0 + 128) * Nw + n0,
                  wt_base + (unsigned)(buf ^ 1) * (128u * 17u * 4u),
                  16, 128, Nw, WFLAGS);
      __builtin_amdgcn_s_wait_tensorcnt(1);
    } else {
      __builtin_amdgcn_s_wait_tensorcnt(0);
    }
    const float* W = &wt[buf][0];
#pragma unroll
    for (int sub = 0; sub < 4; ++sub) {
      const int kk0 = k0 + sub * 32;
      v16h a, bfr;
#pragma unroll
      for (int e = 0; e < 16; ++e) {
        const int ka = 16 * hi + e;
        a[e] = (_Float16)A[m * DMODEL + kk0 + ka];
        bfr[e] = (_Float16)W[(sub * 32 + 16 * hi + e) * 17 + m];
      }
      acc = __builtin_amdgcn_wmma_f32_16x16x32_f16(false, a, false, bfr,
                                                   (short)0, acc, false, false);
    }
  }
#else
  for (int k0 = 0; k0 < DMODEL; k0 += 32) {
    v16h a, bfr;
#pragma unroll
    for (int e = 0; e < 16; ++e) {
      const int ka = 16 * hi + e;
      a[e] = (_Float16)A[m * DMODEL + k0 + ka];
      const int kb = 16 * hi + e;
      bfr[e] = (_Float16)w[(size_t)(k0 + kb) * Nw + n0 + m];
    }
    acc = __builtin_amdgcn_wmma_f32_16x16x32_f16(false, a, false, bfr,
                                                 (short)0, acc, false, false);
  }
#endif
  return acc;
}

// ---------------------------------------------------------------------------
// Kernel 1: fused QK projection (x@wq, x@wk) + RoPE.
// grid = 320 blocks (256 q-tiles + 64 k-tiles), block = 32 (one wave).
// ---------------------------------------------------------------------------
__global__ void k_qkproj(const float* __restrict__ x,
                         const float* __restrict__ wq,
                         const float* __restrict__ wk,
                         const float* __restrict__ fcos,
                         const float* __restrict__ fsin,
                         float* __restrict__ q_rope,
                         float* __restrict__ k_new) {
  const int lane = threadIdx.x;
  const int m  = lane & 15;
  const int hi = lane >> 4;

  const bool isQ = blockIdx.x < 256;
  const int  n0  = isQ ? blockIdx.x * 16 : (blockIdx.x - 256) * 16;
  const float* __restrict__ w = isQ ? wq : wk;
  const int Nw = isQ ? (NHEADS * DH) : (NKV * DH);

  v8f acc = tile_gemm16(x, w, Nw, n0, lane);

  const int col  = n0 + m;
  const int pair = (col & (DH - 1)) >> 1;
  const float c = fcos[pair];
  const float s = fsin[pair];
#pragma unroll
  for (int v = 0; v < 8; ++v) {
    const int row = v + 8 * hi;
    const float t = acc[v];
    const float o = __shfl_xor(t, 1, 32);
    const float r = ((col & 1) == 0) ? (t * c - o * s) : (o * s + t * c);
    if (isQ) q_rope[row * (NHEADS * DH) + col] = r;
    else     k_new [row * (NKV   * DH) + col] = r;
  }
}

// ---------------------------------------------------------------------------
// Kernel 2: flash-decode attention, V == K (reference quirk), split-K over seq.
// grid = (NSPLIT, NKV, BATCH), block = 32 (one wave).
// ---------------------------------------------------------------------------
__global__ void k_attn(const float* __restrict__ cache_k,
                       const float* __restrict__ q_rope,
                       const float* __restrict__ k_new,
                       float* __restrict__ part_o,
                       float* __restrict__ part_m,
                       float* __restrict__ part_l) {
  __shared__ float kt[2][32 * 132];  // 32 keys x 128 dims (+4 pad) per buffer
  __shared__ float pt[16 * 33];      // 16 rows x 32 keys, stride 33

  const int sp = blockIdx.x;
  const int kv = blockIdx.y;
  const int b  = blockIdx.z;
  const int lane = threadIdx.x;
  const int m  = lane & 15;
  const int hi = lane >> 4;
  const int h0 = kv * REP;

  // Q fragments (scale folded in, rows >= REP zero-padded)
  v16h qa[4];
#pragma unroll
  for (int j = 0; j < 4; ++j) {
#pragma unroll
    for (int e = 0; e < 16; ++e) {
      const int dd = 32 * j + 16 * hi + e;
      qa[j][e] = (m < REP)
          ? (_Float16)(q_rope[b * (NHEADS * DH) + (h0 + m) * DH + dd] * QK_SCALE)
          : (_Float16)0.0f;
    }
  }

  float m_run[8], l_run[8];
  v8f acc[8] = {};
#pragma unroll
  for (int v = 0; v < 8; ++v) { m_run[v] = -1e30f; l_run[v] = 0.0f; }

  const int sbase0 = sp * SPLIT_LEN;
  const int NCH = SPLIT_LEN / 32;

#if HAVE_TDM
  const unsigned kt_base = lds_offset_of(&kt[0][0]);
  // data_size=4B | pad_enable | pad_interval=128dw(code 6) | pad_amount=4dw(code 3)
  const int KFLAGS = 0x00020000 | (1 << 20) | (6 << 22) | (3 << 25);
  tdm_load_2d(cache_k + (((size_t)b * SEQ + sbase0) * NKV + kv) * DH,
              kt_base, 128, 32, NKV * DH, KFLAGS);
#elif HAVE_ASYNC
  auto async_issue = [&](int buf, int sb) {
    for (int it = lane; it < 32 * 32; it += 32) {
      const int r  = it >> 5;
      const int f4 = (it & 31) * 4;
      const int s  = sb + r;
      const float* src = (s == LASTPOS)
          ? (k_new + ((size_t)b * NKV + kv) * DH + f4)
          : (cache_k + (((size_t)b * SEQ + s) * NKV + kv) * DH + f4);
      __builtin_amdgcn_global_load_async_to_lds_b128(
          (const __attribute__((address_space(1))) void*)src,
          (__attribute__((address_space(3))) void*)&kt[buf][r * 132 + f4], 0, 0);
    }
  };
  async_issue(0, sbase0);
#endif

  for (int cchunk = 0; cchunk < NCH; ++cchunk) {
    const int sb = sbase0 + cchunk * 32;
    float* K;

#if HAVE_TDM
    if (cchunk + 1 < NCH) {
      tdm_load_2d(cache_k + (((size_t)b * SEQ + sb + 32) * NKV + kv) * DH,
                  kt_base + (unsigned)((cchunk + 1) & 1) * (32u * 132u * 4u),
                  128, 32, NKV * DH, KFLAGS);
      __builtin_amdgcn_s_wait_tensorcnt(1);
    } else {
      __builtin_amdgcn_s_wait_tensorcnt(0);
    }
    K = &kt[cchunk & 1][0];
    if (sb + 31 == LASTPOS) {               // virtual cache update @4095
      *(float4*)&K[31 * 132 + lane * 4] =
          *(const float4*)&k_new[((size_t)b * NKV + kv) * DH + lane * 4];
    }
    __syncthreads();
#elif HAVE_ASYNC
    if (cchunk + 1 < NCH) {
      async_issue((cchunk + 1) & 1, sb + 32);
      asm volatile("s_wait_asynccnt 0x20" ::: "memory");  // chunk cchunk done
    } else {
      asm volatile("s_wait_asynccnt 0x0" ::: "memory");
    }
    K = &kt[cchunk & 1][0];
    __syncthreads();
#else
    __syncthreads();
    K = &kt[0][0];
    for (int it = lane; it < 32 * 32; it += 32) {
      const int r  = it >> 5;
      const int f4 = (it & 31) * 4;
      const int s  = sb + r;
      const float* src = (s == LASTPOS)
          ? (k_new + ((size_t)b * NKV + kv) * DH + f4)
          : (cache_k + (((size_t)b * SEQ + s) * NKV + kv) * DH + f4);
      *(float4*)&K[r * 132 + f4] = *(const float4*)src;
    }
    __syncthreads();
#endif

    // scores: two 16-key tiles, 4 WMMAs each over d
    v8f c0 = {}, c1 = {};
#pragma unroll
    for (int j = 0; j < 4; ++j) {
      v16h kb0, kb1;
#pragma unroll
      for (int e = 0; e < 16; ++e) {
        const int dd = 32 * j + 16 * hi + e;
        kb0[e] = (_Float16)K[m * 132 + dd];
        kb1[e] = (_Float16)K[(16 + m) * 132 + dd];
      }
      c0 = __builtin_amdgcn_wmma_f32_16x16x32_f16(false, qa[j], false, kb0,
                                                  (short)0, c0, false, false);
      c1 = __builtin_amdgcn_wmma_f32_16x16x32_f16(false, qa[j], false, kb1,
                                                  (short)0, c1, false, false);
    }

    // online softmax
    float mloc[8];
#pragma unroll
    for (int v = 0; v < 8; ++v) mloc[v] = fmaxf(c0[v], c1[v]);
#pragma unroll
    for (int off = 1; off < 16; off <<= 1) {
#pragma unroll
      for (int v = 0; v < 8; ++v)
        mloc[v] = fmaxf(mloc[v], __shfl_xor(mloc[v], off, 32));
    }

    float p0[8], p1[8], rsum[8], fac[8];
#pragma unroll
    for (int v = 0; v < 8; ++v) {
      const float mn = fmaxf(m_run[v], mloc[v]);
      fac[v] = __expf(m_run[v] - mn);
      m_run[v] = mn;
      p0[v] = __expf(c0[v] - mn);
      p1[v] = __expf(c1[v] - mn);
      rsum[v] = p0[v] + p1[v];
    }
#pragma unroll
    for (int off = 1; off < 16; off <<= 1) {
#pragma unroll
      for (int v = 0; v < 8; ++v) rsum[v] += __shfl_xor(rsum[v], off, 32);
    }
#pragma unroll
    for (int v = 0; v < 8; ++v) l_run[v] = l_run[v] * fac[v] + rsum[v];
#pragma unroll
    for (int dt = 0; dt < 8; ++dt)
#pragma unroll
      for (int v = 0; v < 8; ++v) acc[dt][v] *= fac[v];

    // re-fragment P via LDS (C-layout -> A-layout)
#pragma unroll
    for (int v = 0; v < 8; ++v) {
      const int row = v + 8 * hi;
      pt[row * 33 + m]      = p0[v];
      pt[row * 33 + 16 + m] = p1[v];
    }
    __syncthreads();
    v16h pa;
#pragma unroll
    for (int e = 0; e < 16; ++e) {
      const int kk = 16 * hi + e;
      pa[e] = (_Float16)pt[m * 33 + kk];
    }

    // out += P * K (reference attends over keys)
#pragma unroll
    for (int dt = 0; dt < 8; ++dt) {
      v16h vb;
#pragma unroll
      for (int e = 0; e < 16; ++e)
        vb[e] = (_Float16)K[(16 * hi + e) * 132 + 16 * dt + m];
      acc[dt] = __builtin_amdgcn_wmma_f32_16x16x32_f16(false, pa, false, vb,
                                                       (short)0, acc[dt],
                                                       false, false);
    }
  }

  if (hi == 0) {
#pragma unroll
    for (int r = 0; r < REP; ++r) {
      const int h = h0 + r;
      const size_t base = ((size_t)(b * NHEADS + h) * NSPLIT + sp) * DH;
#pragma unroll
      for (int dt = 0; dt < 8; ++dt) part_o[base + dt * 16 + m] = acc[dt][r];
      if (m == 0) {
        part_m[(b * NHEADS + h) * NSPLIT + sp] = m_run[r];
        part_l[(b * NHEADS + h) * NSPLIT + sp] = l_run[r];
      }
    }
  }
}

// ---------------------------------------------------------------------------
// Kernel 3: combine split partials. grid = B*H, block = DH.
// ---------------------------------------------------------------------------
__global__ void k_combine(const float* __restrict__ part_o,
                          const float* __restrict__ part_m,
                          const float* __restrict__ part_l,
                          float* __restrict__ attn_out) {
  const int bh = blockIdx.x;
  const int d  = threadIdx.x;
  float mmax = -1e30f;
#pragma unroll
  for (int sp = 0; sp < NSPLIT; ++sp)
    mmax = fmaxf(mmax, part_m[bh * NSPLIT + sp]);
  float lt = 0.0f, o = 0.0f;
#pragma unroll
  for (int sp = 0; sp < NSPLIT; ++sp) {
    const float w = __expf(part_m[bh * NSPLIT + sp] - mmax);
    lt += part_l[bh * NSPLIT + sp] * w;
    o  += part_o[((size_t)bh * NSPLIT + sp) * DH + d] * w;
  }
  attn_out[(size_t)bh * DH + d] = o / lt;
}

// ---------------------------------------------------------------------------
// Kernel 4: output projection attn_out(16x4096) @ wo(4096x4096).
// ---------------------------------------------------------------------------
__global__ void k_oproj(const float* __restrict__ attn_out,
                        const float* __restrict__ wo,
                        float* __restrict__ out) {
  const int lane = threadIdx.x;
  const int m  = lane & 15;
  const int hi = lane >> 4;
  const int n0 = blockIdx.x * 16;

  v8f acc = tile_gemm16(attn_out, wo, DMODEL, n0, lane);

#pragma unroll
  for (int v = 0; v < 8; ++v)
    out[(size_t)(v + 8 * hi) * DMODEL + n0 + m] = acc[v];
}

// ---------------------------------------------------------------------------
extern "C" void kernel_launch(void* const* d_in, const int* in_sizes, int n_in,
                              void* d_out, int out_size, void* d_ws, size_t ws_size,
                              hipStream_t stream) {
  const float* x       = (const float*)d_in[0];
  const float* cache_k = (const float*)d_in[1];
  // d_in[2] cache_v: dead w.r.t. returned tensor (reference attends over keys)
  const float* wq      = (const float*)d_in[3];
  const float* wk      = (const float*)d_in[4];
  // d_in[5] wv: dead
  const float* wo      = (const float*)d_in[6];
  const float* fcos    = (const float*)d_in[7];
  const float* fsin    = (const float*)d_in[8];
  // d_in[9] start_pos == 4095 (fixed by reference shapes)

  float* ws       = (float*)d_ws;
  float* q_rope   = ws;                 // 65536
  float* k_new    = ws + 65536;         // 16384
  float* part_o   = ws + 81920;         // 262144
  float* part_m   = ws + 344064;        // 2048
  float* part_l   = ws + 346112;        // 2048
  float* attn_out = ws + 348160;        // 65536

  k_qkproj<<<320, 32, 0, stream>>>(x, wq, wk, fcos, fsin, q_rope, k_new);

  dim3 g2(NSPLIT, NKV, BATCH);
  k_attn<<<g2, 32, 0, stream>>>(cache_k, q_rope, k_new, part_o, part_m, part_l);

  k_combine<<<BATCH * NHEADS, DH, 0, stream>>>(part_o, part_m, part_l, attn_out);

  k_oproj<<<256, 32, 0, stream>>>(attn_out, wo, (float*)d_out);
}
